// LSTMAutoEncoder_68066641707873
// MI455X (gfx1250) — matrix operand order repaired and checked by reading, not compiled, not run on hardware
//
#include <hip/hip_runtime.h>

typedef __bf16 bf16;
typedef __attribute__((ext_vector_type(16))) __bf16 v16bf;
typedef __attribute__((ext_vector_type(8)))  __bf16 v8bf;
typedef __attribute__((ext_vector_type(8)))  float  v8f;

#define HH 1024
#define BB 32

// ---------------- elementwise helpers ----------------

__global__ void k_cvt_bf16(const float* __restrict__ in, bf16* __restrict__ out, int n) {
  int i = blockIdx.x * blockDim.x + threadIdx.x;
  int stride = gridDim.x * blockDim.x;
  for (; i < n; i += stride) out[i] = (bf16)in[i];
}

__global__ void k_vec_add(const float* __restrict__ a, const float* __restrict__ b,
                          float* __restrict__ out, int n) {
  int i = blockIdx.x * blockDim.x + threadIdx.x;
  int stride = gridDim.x * blockDim.x;
  for (; i < n; i += stride) out[i] = a[i] + b[i];
}

__global__ void k_zero_f32(float* __restrict__ p, int n) {
  int i = blockIdx.x * blockDim.x + threadIdx.x;
  int stride = gridDim.x * blockDim.x;
  for (; i < n; i += stride) p[i] = 0.0f;
}

__global__ void k_copy_int(const int* __restrict__ in, int* __restrict__ out, int n) {
  int i = blockIdx.x * blockDim.x + threadIdx.x;
  if (i < n) out[i] = in[i];
}

// Gather embedding rows (fp32 table -> bf16 activations)
__global__ void k_embed_rows(const int* __restrict__ tokens, const float* __restrict__ emb,
                             bf16* __restrict__ out, int rows, int E) {
  int i = blockIdx.x * blockDim.x + threadIdx.x;
  int n = rows * E;
  int stride = gridDim.x * blockDim.x;
  for (; i < n; i += stride) {
    int r = i / E, e = i - r * E;
    out[i] = (bf16)emb[(long)tokens[r] * E + e];
  }
}

// ---------------- fragment helpers (contiguous-k order, not the ISA table order) ----------------

// A fragment (16x32 bf16): lane -> row m0+(lane&15); K-phase kb=(lane>=16)*16;
// elems 0-15 = K kb..kb+15 (contiguous)  => two 16B loads.
__device__ __forceinline__ v16bf load_a(const bf16* aRow, int kk) {
  v8bf lo = *(const v8bf*)(aRow + kk);
  v8bf hi = *(const v8bf*)(aRow + kk + 8);
  return __builtin_shufflevector(lo, hi,
      0, 1, 2, 3, 4, 5, 6, 7, 8, 9, 10, 11, 12, 13, 14, 15);
}

// ---------------- generic WMMA GEMM: Out[M,N] = X@W^T (+bias) ----------------
// One wave per 16x16 tile; two split-K accumulators for ILP. K multiple of 64.

__global__ void __launch_bounds__(256)
k_gemm_bf16(const bf16* __restrict__ X, int ldx,
            const bf16* __restrict__ W, int ldw,
            float* __restrict__ Out, int ldo,
            const float* __restrict__ bias,
            int M, int N, int K)
{
  const int lane = threadIdx.x & 31;
  const int wv   = threadIdx.x >> 5;
  const int ntn  = N >> 4;
  const int tiles = (M >> 4) * ntn;
  const int tile = blockIdx.x * (blockDim.x >> 5) + wv;
  if (tile >= tiles) return;               // whole-wave exit: EXEC stays all-ones
  const int mt = tile / ntn;
  const int nt = tile - mt * ntn;
  const int m0 = mt << 4, n0 = nt << 4;

  const int lm = lane & 15;
  const int lh = lane >> 4;
  const int cm = m0 + (lh << 3);
  const int cn = n0 + lm;

  v8f c0, c1;
  {
    float bv = bias ? bias[cn] : 0.0f;
#pragma unroll
    for (int r = 0; r < 8; ++r) { c0[r] = bv; c1[r] = 0.0f; }
  }

  const bf16* aRow = X + (long)(m0 + lm) * ldx + (lh << 4);
  const bf16* bRow = W + (long)(n0 + lm) * ldw + (lh << 4);

  for (int kk = 0; kk < K; kk += 64) {
    if (kk + 64 < K) __builtin_prefetch(bRow + kk + 64, 0, 1);
    v16bf a0 = load_a(aRow, kk);
    v16bf b0 = *(const v16bf*)(bRow + kk);
    v16bf a1 = load_a(aRow, kk + 32);
    v16bf b1 = *(const v16bf*)(bRow + kk + 32);
    c0 = __builtin_amdgcn_wmma_f32_16x16x32_bf16(false, a0, false, b0,
                                                 (short)0, c0, false, false);
    c1 = __builtin_amdgcn_wmma_f32_16x16x32_bf16(false, a1, false, b1,
                                                 (short)0, c1, false, false);
  }

#pragma unroll
  for (int r = 0; r < 8; ++r) Out[(long)(cm + r) * ldo + cn] = c0[r] + c1[r];
}

// ---------------- fused LSTM step ----------------
// gates = [pre | bias] + x@wih^T + hprev@whh^T ; then pointwise -> c, hout.
// One wave owns a 16x16 (batch x j) tile and all 4 gate tiles (shared A, 4
// independent WMMA chains). hprev (32x1024 bf16 = 64KB) is staged into LDS via
// global_load_async_to_lds_b128 (ASYNCcnt) and A-fragments read via ds loads.
// Fixed shape: Bn=32, H=1024 -> 128 wave-tiles = 16 blocks x 8 waves.

__global__ void __launch_bounds__(256)
k_lstm_step(const bf16* __restrict__ x, const bf16* __restrict__ wih, int K1,
            const bf16* __restrict__ hprev, const bf16* __restrict__ whh,
            const float* __restrict__ pre, const float* __restrict__ bias,
            float* __restrict__ cst, bf16* __restrict__ hout,
            bf16* __restrict__ hist)
{
  __shared__ __align__(16) bf16 shA[BB * HH];       // 64KB previous-h tile

  // ---- async stage hprev -> LDS (each lane: 16B chunks) ----
  {
    const unsigned base = (unsigned)(unsigned long long)(void*)shA;
    const char* g = (const char*)hprev;
    const int tid = threadIdx.x;
#pragma unroll
    for (int it = 0; it < (BB * HH * 2) / (256 * 16); ++it) {  // 16 iters
      const int off = (it * 256 + tid) * 16;
      unsigned lds_addr = base + (unsigned)off;
      unsigned long long gaddr = (unsigned long long)(g + off);
      asm volatile("global_load_async_to_lds_b128 %0, %1, off"
                   :: "v"(lds_addr), "v"(gaddr) : "memory");
    }
    asm volatile("s_wait_asynccnt 0" ::: "memory");
    __syncthreads();
  }

  const int lane = threadIdx.x & 31;
  const int wv   = threadIdx.x >> 5;
  const int tile = blockIdx.x * 8 + wv;             // 128 tiles: 2 x 64
  const int mt = tile >> 6;
  const int nt = tile & 63;
  const int m0 = mt << 4, n0 = nt << 4;

  const int lm = lane & 15;
  const int lh = lane >> 4;
  const int cm = m0 + (lh << 3);                    // batch row base
  const int cn = n0 + lm;                           // j column

  v8f acc[4];
  if (pre) {
#pragma unroll
    for (int gt = 0; gt < 4; ++gt)
#pragma unroll
      for (int r = 0; r < 8; ++r)
        acc[gt][r] = pre[(long)(cm + r) * (4 * HH) + gt * HH + cn];
  } else {
#pragma unroll
    for (int gt = 0; gt < 4; ++gt) {
      float bv = bias[gt * HH + cn];
#pragma unroll
      for (int r = 0; r < 8; ++r) acc[gt][r] = bv;
    }
  }

  // optional input GEMM: x[BB,K1] @ wih[4H,K1]^T
  if (x) {
    const bf16* aRow = x + (long)(m0 + lm) * K1 + (lh << 4);
    const bf16* bR0 = wih + (long)(0 * HH + n0 + lm) * K1 + (lh << 4);
    const bf16* bR1 = wih + (long)(1 * HH + n0 + lm) * K1 + (lh << 4);
    const bf16* bR2 = wih + (long)(2 * HH + n0 + lm) * K1 + (lh << 4);
    const bf16* bR3 = wih + (long)(3 * HH + n0 + lm) * K1 + (lh << 4);
    for (int kk = 0; kk < K1; kk += 32) {
      v16bf a = load_a(aRow, kk);
      v16bf b0 = *(const v16bf*)(bR0 + kk);
      v16bf b1 = *(const v16bf*)(bR1 + kk);
      v16bf b2 = *(const v16bf*)(bR2 + kk);
      v16bf b3 = *(const v16bf*)(bR3 + kk);
      acc[0] = __builtin_amdgcn_wmma_f32_16x16x32_bf16(false, a, false, b0, (short)0, acc[0], false, false);
      acc[1] = __builtin_amdgcn_wmma_f32_16x16x32_bf16(false, a, false, b1, (short)0, acc[1], false, false);
      acc[2] = __builtin_amdgcn_wmma_f32_16x16x32_bf16(false, a, false, b2, (short)0, acc[2], false, false);
      acc[3] = __builtin_amdgcn_wmma_f32_16x16x32_bf16(false, a, false, b3, (short)0, acc[3], false, false);
    }
  }

  // recurrent GEMM: hprev(LDS) @ whh[4H,H]^T
  {
    const bf16* aRow = shA + (m0 + lm) * HH + (lh << 4);
    const bf16* bR0 = whh + (long)(0 * HH + n0 + lm) * HH + (lh << 4);
    const bf16* bR1 = whh + (long)(1 * HH + n0 + lm) * HH + (lh << 4);
    const bf16* bR2 = whh + (long)(2 * HH + n0 + lm) * HH + (lh << 4);
    const bf16* bR3 = whh + (long)(3 * HH + n0 + lm) * HH + (lh << 4);
    for (int kk = 0; kk < HH; kk += 32) {
      if (kk + 32 < HH) __builtin_prefetch(bR0 + kk + 32, 0, 1);
      v16bf a = load_a(aRow, kk);                 // ds loads from LDS
      v16bf b0 = *(const v16bf*)(bR0 + kk);
      v16bf b1 = *(const v16bf*)(bR1 + kk);
      v16bf b2 = *(const v16bf*)(bR2 + kk);
      v16bf b3 = *(const v16bf*)(bR3 + kk);
      acc[0] = __builtin_amdgcn_wmma_f32_16x16x32_bf16(false, a, false, b0, (short)0, acc[0], false, false);
      acc[1] = __builtin_amdgcn_wmma_f32_16x16x32_bf16(false, a, false, b1, (short)0, acc[1], false, false);
      acc[2] = __builtin_amdgcn_wmma_f32_16x16x32_bf16(false, a, false, b2, (short)0, acc[2], false, false);
      acc[3] = __builtin_amdgcn_wmma_f32_16x16x32_bf16(false, a, false, b3, (short)0, acc[3], false, false);
    }
  }

  // pointwise LSTM update in registers (PyTorch gate order i,f,g,o)
#pragma unroll
  for (int r = 0; r < 8; ++r) {
    const long idx = (long)(cm + r) * HH + cn;
    float gi = acc[0][r], gf = acc[1][r], gg = acc[2][r], go = acc[3][r];
    float si = 1.0f / (1.0f + __expf(-gi));
    float sf = 1.0f / (1.0f + __expf(-gf));
    float so = 1.0f / (1.0f + __expf(-go));
    float cnew = sf * cst[idx] + si * tanhf(gg);
    float hn = so * tanhf(cnew);
    cst[idx] = cnew;
    bf16 hb = (bf16)hn;
    hout[idx] = hb;
    if (hist) hist[idx] = hb;
  }
}

// ---------------- argmax over V + teacher-forcing select ----------------

__global__ void __launch_bounds__(256)
k_argmax_select(const float* __restrict__ pred, const int* __restrict__ src_t,
                const unsigned char* __restrict__ tf_mask, int t,
                int* __restrict__ tok, int V)
{
  const int b = blockIdx.x;
  const float* row = pred + (long)b * V;
  float best = -__builtin_inff();
  int bidx = 0x7fffffff;
  for (int i = threadIdx.x; i < V; i += blockDim.x) {
    float v = row[i];
    if (v > best || (v == best && i < bidx)) { best = v; bidx = i; }
  }
  __shared__ float sv[256];
  __shared__ int   si[256];
  sv[threadIdx.x] = best; si[threadIdx.x] = bidx;
  __syncthreads();
  for (int s = blockDim.x >> 1; s > 0; s >>= 1) {
    if ((int)threadIdx.x < s) {
      float ov = sv[threadIdx.x + s]; int oi = si[threadIdx.x + s];
      if (ov > sv[threadIdx.x] || (ov == sv[threadIdx.x] && oi < si[threadIdx.x])) {
        sv[threadIdx.x] = ov; si[threadIdx.x] = oi;
      }
    }
    __syncthreads();
  }
  if (threadIdx.x == 0) tok[b] = tf_mask[t] ? src_t[b] : si[0];
}

// ---------------- host-side orchestration ----------------

extern "C" void kernel_launch(void* const* d_in, const int* in_sizes, int n_in,
                              void* d_out, int out_size, void* d_ws, size_t ws_size,
                              hipStream_t stream)
{
  const int V = 32000, E = 128, H = 1024, T = 64, Bn = 32;
  const int H4 = 4 * H, TB = T * Bn;

  const int*           source  = (const int*)d_in[0];
  const unsigned char* tf_mask = (const unsigned char*)d_in[1];
  const float* emb_enc = (const float*)d_in[2];
  const float* emb_dec = (const float*)d_in[3];
  const float* fc_w    = (const float*)d_in[4];
  const float* fc_b    = (const float*)d_in[5];
  const float* e_wih0 = (const float*)d_in[6],  *e_whh0 = (const float*)d_in[7];
  const float* e_bih0 = (const float*)d_in[8],  *e_bhh0 = (const float*)d_in[9];
  const float* e_wih1 = (const float*)d_in[10], *e_whh1 = (const float*)d_in[11];
  const float* e_bih1 = (const float*)d_in[12], *e_bhh1 = (const float*)d_in[13];
  const float* d_wih0 = (const float*)d_in[14], *d_whh0 = (const float*)d_in[15];
  const float* d_bih0 = (const float*)d_in[16], *d_bhh0 = (const float*)d_in[17];
  const float* d_wih1 = (const float*)d_in[18], *d_whh1 = (const float*)d_in[19];
  const float* d_bih1 = (const float*)d_in[20], *d_bhh1 = (const float*)d_in[21];

  // ---- workspace layout ----
  char* ws = (char*)d_ws;
  size_t off = 0;
  auto alloc = [&](size_t bytes) -> char* {
    char* p = ws + off;
    off += (bytes + 255) & ~(size_t)255;
    return p;
  };
  bf16* fcw_bf    = (bf16*)alloc((size_t)V * H * 2);
  bf16* e_wih0_bf = (bf16*)alloc((size_t)H4 * E * 2);
  bf16* e_whh0_bf = (bf16*)alloc((size_t)H4 * H * 2);
  bf16* e_wih1_bf = (bf16*)alloc((size_t)H4 * H * 2);
  bf16* e_whh1_bf = (bf16*)alloc((size_t)H4 * H * 2);
  bf16* d_wih0_bf = (bf16*)alloc((size_t)H4 * E * 2);
  bf16* d_whh0_bf = (bf16*)alloc((size_t)H4 * H * 2);
  bf16* d_wih1_bf = (bf16*)alloc((size_t)H4 * H * 2);
  bf16* d_whh1_bf = (bf16*)alloc((size_t)H4 * H * 2);
  bf16* xe_bf     = (bf16*)alloc((size_t)TB * E * 2);
  bf16* ys0_bf    = (bf16*)alloc((size_t)TB * H * 2);
  float* gates_all = (float*)alloc((size_t)TB * H4 * 4);
  float* b_e0 = (float*)alloc((size_t)H4 * 4);
  float* b_e1 = (float*)alloc((size_t)H4 * 4);
  float* b_d0 = (float*)alloc((size_t)H4 * 4);
  float* b_d1 = (float*)alloc((size_t)H4 * 4);
  float* c0   = (float*)alloc((size_t)Bn * H * 4);
  float* c1   = (float*)alloc((size_t)Bn * H * 4);
  bf16*  h0A = (bf16*)alloc((size_t)Bn * H * 2);
  bf16*  h0B = (bf16*)alloc((size_t)Bn * H * 2);
  bf16*  h1A = (bf16*)alloc((size_t)Bn * H * 2);
  bf16*  h1B = (bf16*)alloc((size_t)Bn * H * 2);
  bf16*  x_bf = (bf16*)alloc((size_t)Bn * E * 2);
  int*   tok  = (int*)alloc((size_t)Bn * 4);
  (void)ws_size; (void)in_sizes; (void)n_in; (void)out_size;

  auto cvt = [&](const float* src, bf16* dst, long n) {
    int blocks = (int)((n + 255) / 256); if (blocks > 2048) blocks = 2048;
    k_cvt_bf16<<<blocks, 256, 0, stream>>>(src, dst, (int)n);
  };
  auto vadd = [&](const float* a, const float* b, float* o, int n) {
    k_vec_add<<<(n + 255) / 256, 256, 0, stream>>>(a, b, o, n);
  };
  auto zero = [&](float* p, long n) {
    int blocks = (int)((n + 255) / 256); if (blocks > 2048) blocks = 2048;
    k_zero_f32<<<blocks, 256, 0, stream>>>(p, (int)n);
  };
  auto gemm = [&](const bf16* X, int ldx, const bf16* W, int ldw,
                  float* Out, int ldo, const float* bias, int M, int N, int K) {
    int tiles = (M >> 4) * (N >> 4);
    int blocks = (tiles + 7) / 8;
    k_gemm_bf16<<<blocks, 256, 0, stream>>>(X, ldx, W, ldw, Out, ldo, bias, M, N, K);
  };
  auto step = [&](const bf16* x, const bf16* wih, int K1,
                  const bf16* hprev, const bf16* whh,
                  const float* pre, const float* bias,
                  float* c, bf16* hout, bf16* hist) {
    k_lstm_step<<<16, 256, 0, stream>>>(x, wih, K1, hprev, whh, pre, bias, c, hout, hist);
  };

  // ---- one-time: weights -> bf16 (resident in 192MB L2 afterwards) ----
  cvt(fc_w,   fcw_bf,    (long)V * H);
  cvt(e_wih0, e_wih0_bf, (long)H4 * E);
  cvt(e_whh0, e_whh0_bf, (long)H4 * H);
  cvt(e_wih1, e_wih1_bf, (long)H4 * H);
  cvt(e_whh1, e_whh1_bf, (long)H4 * H);
  cvt(d_wih0, d_wih0_bf, (long)H4 * E);
  cvt(d_whh0, d_whh0_bf, (long)H4 * H);
  cvt(d_wih1, d_wih1_bf, (long)H4 * H);
  cvt(d_whh1, d_whh1_bf, (long)H4 * H);
  vadd(e_bih0, e_bhh0, b_e0, H4);
  vadd(e_bih1, e_bhh1, b_e1, H4);
  vadd(d_bih0, d_bhh0, b_d0, H4);
  vadd(d_bih1, d_bhh1, b_d1, H4);

  zero(c0, (long)Bn * H);
  zero(c1, (long)Bn * H);
  zero((float*)h0A, (long)Bn * H / 2);     // bf16 zeros == f32 zeros bitwise
  zero((float*)h1A, (long)Bn * H / 2);
  zero((float*)d_out, (long)Bn * V);       // outputs[0] stays zero

  bf16 *h0cur = h0A, *h0nxt = h0B, *h1cur = h1A, *h1nxt = h1B;

  // ---- encoder ----
  k_embed_rows<<<(TB * E + 255) / 256, 256, 0, stream>>>(source, emb_enc, xe_bf, TB, E);
  // layer 0: batched input GEMM over all T, then fused recurrent step per t
  gemm(xe_bf, E, e_wih0_bf, E, gates_all, H4, b_e0, TB, H4, E);
  for (int t = 0; t < T; ++t) {
    step(nullptr, nullptr, 0, h0cur, e_whh0_bf,
         gates_all + (size_t)t * Bn * H4, nullptr,
         c0, h0nxt, ys0_bf + (size_t)t * Bn * H);
    bf16* tmp = h0cur; h0cur = h0nxt; h0nxt = tmp;
  }
  // layer 1
  gemm(ys0_bf, H, e_wih1_bf, H, gates_all, H4, b_e1, TB, H4, H);
  for (int t = 0; t < T; ++t) {
    step(nullptr, nullptr, 0, h1cur, e_whh1_bf,
         gates_all + (size_t)t * Bn * H4, nullptr, c1, h1nxt, nullptr);
    bf16* tmp = h1cur; h1cur = h1nxt; h1nxt = tmp;
  }

  // ---- autoregressive decoder with teacher forcing ----
  k_copy_int<<<1, Bn, 0, stream>>>(source, tok, Bn);   // x0 = source[0]
  for (int t = 1; t < T; ++t) {
    k_embed_rows<<<(Bn * E + 255) / 256, 256, 0, stream>>>(tok, emb_dec, x_bf, Bn, E);
    step(x_bf, d_wih0_bf, E, h0cur, d_whh0_bf, nullptr, b_d0, c0, h0nxt, nullptr);
    { bf16* tmp = h0cur; h0cur = h0nxt; h0nxt = tmp; }
    step(h0cur, d_wih1_bf, H, h1cur, d_whh1_bf, nullptr, b_d1, c1, h1nxt, nullptr);
    { bf16* tmp = h1cur; h1cur = h1nxt; h1nxt = tmp; }
    float* pred = (float*)d_out + (size_t)t * Bn * V;
    gemm(h1cur, H, fcw_bf, H, pred, V, fc_b, Bn, V, H);   // 32x32000x1024 FC
    k_argmax_select<<<Bn, 256, 0, stream>>>(pred, source + (size_t)t * Bn,
                                            tf_mask, t, tok, V);
  }
}
